// Block_29008209117188
// MI455X (gfx1250) — matrix operand order repaired and checked by reading, not compiled, not run on hardware
//
#include <hip/hip_runtime.h>
#include <hip/hip_bf16.h>
#include <math.h>

// ---------------- problem constants ----------------
#define BATCH   8
#define SEQ     8192
#define DIM     256
#define HEADS   8
#define HD      32
#define GHEADS  7          // global (linear-attn) heads, heads 1..7
#define WINDOW  64
#define MLPH    1024
#define ROWS    (BATCH * SEQ)   // 65536
#define EPSLN   1e-5f
#define SCALE_HD 0.17677669529663687f   // 32^-0.5

typedef _Float16 half_t;
typedef __attribute__((ext_vector_type(16))) _Float16 v16h;
typedef __attribute__((ext_vector_type(8)))  _Float16 v8h;
typedef __attribute__((ext_vector_type(8)))  float    v8f;
typedef unsigned int u32;
typedef __attribute__((ext_vector_type(4))) unsigned int u32x4;
typedef __attribute__((ext_vector_type(8))) unsigned int u32x8;

// LDS B-tile row stride (halves): 256 data + 8 pad = 264 halves = 132 dwords.
// 132 % 64 == 4  -> row r starts at bank 4r: conflict-friendly for ds_load_b128.
#define BROW 264

// ---------------- WMMA helpers (CDNA5 wave32) ----------------
// A fragment: lane r=lane%16 holds row r; group g=lane/16 holds K in
// {8g..8g+7} U {16+8g..16+8g+7}  (ISA 7.12.2, 16-bit A 16x32)
__device__ __forceinline__ v16h load_a_frag(const half_t* __restrict__ p, int ld) {
    int lane = threadIdx.x & 31;
    int r = lane & 15, g = lane >> 4;
    const half_t* row = p + (size_t)r * ld + 8 * g;
    v8h lo = *(const v8h*)(row);
    v8h hi = *(const v8h*)(row + 16);
    return __builtin_shufflevector(lo, hi, 0,1,2,3,4,5,6,7,8,9,10,11,12,13,14,15);
}
// B fragment (B = W^T, W row-major [N][K]): lane r holds W row (col of B);
// group g holds K in {8g..8g+7} U {16+8g..16+8g+7} (same order as A).
__device__ __forceinline__ v16h load_b_frag(const half_t* p, int ld) {
    int lane = threadIdx.x & 31;
    int r = lane & 15, g = lane >> 4;
    const half_t* row = p + (size_t)r * ld + 8 * g;
    v8h lo = *(const v8h*)(row);
    v8h hi = *(const v8h*)(row + 16);
    return __builtin_shufflevector(lo, hi, 0,1,2,3,4,5,6,7,8,9,10,11,12,13,14,15);
}
__device__ __forceinline__ v8f wmma16(v16h a, v16h b, v8f c) {
    return __builtin_amdgcn_wmma_f32_16x16x32_f16(false, a, false, b, (short)0, c, false, false);
}

// ---------------- f32 -> f16 weight conversion ----------------
__global__ void cvt_f32_f16(const float* __restrict__ s, half_t* __restrict__ d, int n) {
    int i = blockIdx.x * 256 + threadIdx.x;
    if (i < n) d[i] = (half_t)s[i];
}

// ---------------- LayerNorm (wave per 256-wide row) ----------------
__global__ void ln_kernel(const float* __restrict__ x, const float* __restrict__ gw,
                          const float* __restrict__ bw, half_t* __restrict__ o) {
    int wid = threadIdx.x >> 5, lane = threadIdx.x & 31;
    size_t row = (size_t)blockIdx.x * 8 + wid;
    const float* xr = x + row * DIM;
    float v[8], s = 0.f, ss = 0.f;
#pragma unroll
    for (int i = 0; i < 8; ++i) { v[i] = xr[lane + 32 * i]; s += v[i]; ss += v[i] * v[i]; }
#pragma unroll
    for (int off = 16; off; off >>= 1) { s += __shfl_xor(s, off, 32); ss += __shfl_xor(ss, off, 32); }
    float mean = s * (1.f / DIM);
    float var  = ss * (1.f / DIM) - mean * mean;
    float rstd = rsqrtf(var + EPSLN);
    half_t* orow = o + row * DIM;
#pragma unroll
    for (int i = 0; i < 8; ++i) {
        int c = lane + 32 * i;
        orow[c] = (half_t)((v[i] - mean) * rstd * gw[c] + bw[c]);
    }
}

// ---------------- generic WMMA GEMM:  out[M,N] = A[M,K] @ W[N,K]^T ----------------
// EPI 0: store f16 ; EPI 1: +bias +res, store f32 ; EPI 2: +bias, GELU(exact), store f16
// LDSB: stage the 64-row weight slab in LDS via the Tensor Data Mover (K must be 256).
template <int EPI, bool LDSB>
__global__ void gemm_f16_kernel(const half_t* __restrict__ A, const half_t* __restrict__ W,
                                const float* __restrict__ bias, const float* __restrict__ res,
                                half_t* __restrict__ outH, float* __restrict__ outF,
                                int M, int Nd, int K) {
    __shared__ half_t lds_b[LDSB ? 64 * BROW : 8];

    if constexpr (LDSB) {
        // One TDM 2D-tile load per block: W[n0..n0+63][0..255] (f16) -> LDS,
        // padded +4 dwords every 128 dwords (row stride 132 dwords in LDS).
        if (threadIdx.x < 32) {
            const half_t* wtile = W + (size_t)(blockIdx.y * 64) * K;
            unsigned long long ga = (unsigned long long)(uintptr_t)wtile;
            u32 ldsoff = (u32)(uintptr_t)(&lds_b[0]);   // LDS aperture: offset in addr[31:0]
            u32 k2 = (u32)(K >> 1);                     // dwords per row = 128
            u32x4 g0;
            g0[0] = 1u;                                 // count=1, user descriptor
            g0[1] = ldsoff;                             // lds_addr (bytes)
            g0[2] = (u32)ga;                            // global_addr[31:0]
            g0[3] = ((u32)(ga >> 32) & 0x1FFFFFFu) | 0x80000000u;  // addr[56:32] | type=2
            u32x8 g1;
            g1[0] = 0x07920000u;  // data_size=4B | pad_enable | pad_interval=128dw | pad_amount=4dw
            g1[1] = (k2 & 0xFFFFu) << 16;               // tensor_dim0[15:0]
            g1[2] = ((u32)Nd & 0xFFFFu) << 16;          // tensor_dim0 hi | tensor_dim1[15:0]
            g1[3] = (k2 & 0xFFFFu) << 16;               // tensor_dim1 hi | tile_dim0=128dw
            g1[4] = 64u;                                // tile_dim1=64 rows, tile_dim2=0
            g1[5] = k2;                                 // tensor_dim0_stride lo32 (dwords)
            g1[6] = 0u;
            g1[7] = 0u;
            asm volatile("tensor_load_to_lds %0, %1" :: "s"(g0), "s"(g1) : "memory");
            __builtin_amdgcn_s_wait_tensorcnt(0);
        }
        __syncthreads();
    }

    int wave = threadIdx.x >> 5;
    int wm = wave & 3, wn = wave >> 2;
    int m0 = blockIdx.x * 128 + wm * 32;
    int n0 = blockIdx.y * 64  + wn * 32;
    v8f acc[2][2] = {};
    const half_t* a0p = A + (size_t)m0 * K;
    const half_t* a1p = A + (size_t)(m0 + 16) * K;
    const half_t* b0p;
    const half_t* b1p;
    int bld;
    if constexpr (LDSB) {
        b0p = lds_b + (size_t)(wn * 32) * BROW;
        b1p = b0p + 16 * BROW;
        bld = BROW;
    } else {
        b0p = W + (size_t)n0 * K;
        b1p = W + (size_t)(n0 + 16) * K;
        bld = K;
    }
    for (int k0 = 0; k0 < K; k0 += 32) {
        v16h a0 = load_a_frag(a0p + k0, K);
        v16h a1 = load_a_frag(a1p + k0, K);
        v16h b0 = load_b_frag(b0p + k0, bld);
        v16h b1 = load_b_frag(b1p + k0, bld);
        acc[0][0] = wmma16(a0, b0, acc[0][0]);
        acc[0][1] = wmma16(a0, b1, acc[0][1]);
        acc[1][0] = wmma16(a1, b0, acc[1][0]);
        acc[1][1] = wmma16(a1, b1, acc[1][1]);
    }
    int lane = threadIdx.x & 31, r = lane & 15, g = lane >> 4;
#pragma unroll
    for (int fm = 0; fm < 2; ++fm)
#pragma unroll
    for (int fn = 0; fn < 2; ++fn) {
        int col = n0 + fn * 16 + r;
        float bvl = (EPI >= 1) ? bias[col] : 0.f;
#pragma unroll
        for (int v = 0; v < 8; ++v) {
            int row = m0 + fm * 16 + v + 8 * g;   // C layout: vgpr v, lanes 16-31 are M+8
            float val = acc[fm][fn][v] + bvl;
            size_t idx = (size_t)row * Nd + col;
            if (EPI == 0) {
                outH[idx] = (half_t)val;
            } else if (EPI == 1) {
                outF[idx] = val + res[idx];
            } else {
                float gl = 0.5f * val * (1.f + erff(val * 0.70710678118654752f));
                outH[idx] = (half_t)gl;
            }
        }
    }
}

// ---------------- linear attention: column softmax stats over n ----------------
// one wave per (b, h-1, d): 8*7*32 = 1792 waves
__global__ void lin_kstats(const half_t* __restrict__ kv, float* __restrict__ kmax,
                           float* __restrict__ ksuminv) {
    int wid  = threadIdx.x >> 5, lane = threadIdx.x & 31;
    int widx = blockIdx.x * 8 + wid;
    int b = widx / (GHEADS * HD);
    int rem = widx % (GHEADS * HD);
    int h = rem / HD + 1;
    int d = rem % HD;
    int c = h * HD + d;
    float m = -1e30f, s = 0.f;
    for (int n = lane; n < SEQ; n += 32) {
        float v = (float)kv[((size_t)b * SEQ + n) * DIM + c];
        if (v > m) { s = s * __expf(m - v) + 1.f; m = v; }
        else       { s += __expf(v - m); }
    }
#pragma unroll
    for (int off = 16; off; off >>= 1) {
        float om = __shfl_xor(m, off, 32);
        float os = __shfl_xor(s, off, 32);
        float M  = fmaxf(m, om);
        s = s * __expf(m - M) + os * __expf(om - M);
        m = M;
    }
    if (lane == 0) { kmax[widx] = m; ksuminv[widx] = 1.f / s; }
}

// ---------------- linear attention: ctx[d,e] = sum_n ks[n,d] * v[n,e] ----------------
// one block per (b,h), 1024 threads, thread = (d = tid/32, e = tid%32)
__global__ void lin_ctx(const half_t* __restrict__ kv, const float* __restrict__ kmax,
                        const float* __restrict__ ksuminv, float* __restrict__ ctx) {
    __shared__ float kch[32][32];
    __shared__ float vch[32][32];
    int bh = blockIdx.x;           // 0..55
    int b = bh / GHEADS, hm = bh % GHEADS, h = hm + 1;
    int d = threadIdx.x >> 5, e = threadIdx.x & 31;
    int li = threadIdx.x >> 5, lc = threadIdx.x & 31;   // load mapping
    int statbase = b * (GHEADS * HD) + hm * HD;
    float km = kmax[statbase + lc];
    float ki = ksuminv[statbase + lc];
    float acc = 0.f;
    for (int ch = 0; ch < SEQ / 32; ++ch) {
        int n = ch * 32 + li;
        float v = (float)kv[((size_t)b * SEQ + n) * DIM + h * HD + lc];
        kch[li][lc] = __expf(v - km) * ki;
        vch[li][lc] = v;
        __syncthreads();
#pragma unroll
        for (int i = 0; i < 32; ++i) acc += kch[i][d] * vch[i][e];
        __syncthreads();
    }
    ctx[(size_t)bh * (HD * HD) + d * HD + e] = acc;
}

// ---------------- linear attention: out[n,e] = softmax_d(q) * 32^-0.5 @ ctx ----------------
// one wave per (b,h,n): 56*8192 waves
__global__ void lin_out(const half_t* __restrict__ q, const float* __restrict__ ctx,
                        half_t* __restrict__ attn) {
    int wid  = threadIdx.x >> 5, lane = threadIdx.x & 31;
    int widx = blockIdx.x * 8 + wid;
    int n  = widx & (SEQ - 1);
    int bh = widx >> 13;           // /8192
    int b = bh / GHEADS, hm = bh % GHEADS, h = hm + 1;
    size_t row = (size_t)b * SEQ + n;
    float qv = (float)q[row * DIM + h * HD + lane];
    float m = qv;
#pragma unroll
    for (int off = 16; off; off >>= 1) m = fmaxf(m, __shfl_xor(m, off, 32));
    float e = __expf(qv - m);
    float s = e;
#pragma unroll
    for (int off = 16; off; off >>= 1) s += __shfl_xor(s, off, 32);
    float qs = e / s * SCALE_HD;
    const float* cb = ctx + (size_t)bh * (HD * HD);
    float acc = 0.f;
#pragma unroll
    for (int d = 0; d < 32; ++d) {
        float qsd = __shfl(qs, d, 32);
        acc += qsd * cb[d * HD + lane];
    }
    attn[row * DIM + h * HD + lane] = (half_t)acc;
}

// ---------------- local attention head 0 (window 64, lookaround 3x) ----------------
// one block per (b, window): 8*128 = 1024 blocks, 256 threads
__global__ void local_attn_kernel(const half_t* __restrict__ q, const half_t* __restrict__ kv,
                                  half_t* __restrict__ attn) {
    __shared__ float bq[64][32];
    __shared__ float kvw[192][32];
    __shared__ float dots[64][192];
    __shared__ float pmax[64][4];
    __shared__ float psum[64][4];
    int blk = blockIdx.x;
    int b = blk >> 7, w = blk & 127;
    int t = threadIdx.x;
    size_t qbase = ((size_t)b * SEQ + (size_t)w * WINDOW) * DIM;
    for (int idx = t; idx < 64 * 32; idx += 256) {
        int j = idx >> 5, c = idx & 31;
        bq[j][c] = (float)q[qbase + (size_t)j * DIM + c] * SCALE_HD;
    }
    for (int idx = t; idx < 192 * 32; idx += 256) {
        int j = idx >> 5, c = idx & 31;
        int n = w * WINDOW - WINDOW + j;
        float v = 0.f;
        if (n >= 0 && n < SEQ) v = (float)kv[((size_t)b * SEQ + n) * DIM + c];
        kvw[j][c] = v;
    }
    __syncthreads();
    int r = t >> 2, cg = t & 3;
    float mloc = -1e30f;
    for (int jj = 0; jj < 48; ++jj) {
        int j = cg * 48 + jj;
        int n = w * WINDOW - WINDOW + j;
        float dv;
        if (n >= 0 && n < SEQ) {
            float acc = 0.f;
#pragma unroll
            for (int c = 0; c < 32; ++c) acc += bq[r][c] * kvw[j][c];
            dv = acc;
        } else dv = -1e9f;
        dots[r][j] = dv;
        mloc = fmaxf(mloc, dv);
    }
    pmax[r][cg] = mloc;
    __syncthreads();
    float m = fmaxf(fmaxf(pmax[r][0], pmax[r][1]), fmaxf(pmax[r][2], pmax[r][3]));
    float sloc = 0.f;
    for (int jj = 0; jj < 48; ++jj) {
        int j = cg * 48 + jj;
        float e = __expf(dots[r][j] - m);
        dots[r][j] = e;
        sloc += e;
    }
    psum[r][cg] = sloc;
    __syncthreads();
    float inv = 1.f / (psum[r][0] + psum[r][1] + psum[r][2] + psum[r][3]);
#pragma unroll
    for (int i = 0; i < 8; ++i) {
        int e = cg * 8 + i;
        float acc = 0.f;
        for (int j = 0; j < 192; ++j) acc += dots[r][j] * kvw[j][e];
        attn[qbase + (size_t)r * DIM + e] = (half_t)(acc * inv);
    }
}

// ---------------- launch ----------------
extern "C" void kernel_launch(void* const* d_in, const int* in_sizes, int n_in,
                              void* d_out, int out_size, void* d_ws, size_t ws_size,
                              hipStream_t stream) {
    const float* x      = (const float*)d_in[0];
    const float* n1g    = (const float*)d_in[1];
    const float* n1b    = (const float*)d_in[2];
    const float* Wq     = (const float*)d_in[3];
    const float* Wkv    = (const float*)d_in[4];
    const float* Wproj  = (const float*)d_in[5];
    const float* bproj  = (const float*)d_in[6];
    const float* n2g    = (const float*)d_in[7];
    const float* n2b    = (const float*)d_in[8];
    const float* W1     = (const float*)d_in[9];
    const float* b1     = (const float*)d_in[10];
    const float* W2     = (const float*)d_in[11];
    const float* b2     = (const float*)d_in[12];
    float* out = (float*)d_out;

    char* p = (char*)d_ws;
    auto alloc = [&](size_t bytes) -> char* {
        char* r = p;
        p += (bytes + 255) & ~(size_t)255;
        return r;
    };
    half_t* h16    = (half_t*)alloc((size_t)ROWS * DIM * 2);
    half_t* q16    = (half_t*)alloc((size_t)ROWS * DIM * 2);
    half_t* kv16   = (half_t*)alloc((size_t)ROWS * DIM * 2);
    half_t* attn16 = (half_t*)alloc((size_t)ROWS * DIM * 2);
    half_t* h2_16  = (half_t*)alloc((size_t)ROWS * DIM * 2);
    half_t* mid16  = (half_t*)alloc((size_t)ROWS * MLPH * 2);
    float*  x2     = (float*) alloc((size_t)ROWS * DIM * 4);
    half_t* wq16   = (half_t*)alloc((size_t)DIM * DIM * 2);
    half_t* wkv16  = (half_t*)alloc((size_t)DIM * DIM * 2);
    half_t* wpr16  = (half_t*)alloc((size_t)DIM * DIM * 2);
    half_t* w1_16  = (half_t*)alloc((size_t)MLPH * DIM * 2);
    half_t* w2_16  = (half_t*)alloc((size_t)DIM * MLPH * 2);
    float*  kmax   = (float*) alloc((size_t)BATCH * GHEADS * HD * 4);
    float*  ksinv  = (float*) alloc((size_t)BATCH * GHEADS * HD * 4);
    float*  ctx    = (float*) alloc((size_t)BATCH * GHEADS * HD * HD * 4);

    // weights -> f16
    cvt_f32_f16<<<(DIM * DIM + 255) / 256, 256, 0, stream>>>(Wq,    wq16,  DIM * DIM);
    cvt_f32_f16<<<(DIM * DIM + 255) / 256, 256, 0, stream>>>(Wkv,   wkv16, DIM * DIM);
    cvt_f32_f16<<<(DIM * DIM + 255) / 256, 256, 0, stream>>>(Wproj, wpr16, DIM * DIM);
    cvt_f32_f16<<<(MLPH * DIM + 255) / 256, 256, 0, stream>>>(W1,   w1_16, MLPH * DIM);
    cvt_f32_f16<<<(MLPH * DIM + 255) / 256, 256, 0, stream>>>(W2,   w2_16, MLPH * DIM);

    // LN1
    ln_kernel<<<ROWS / 8, 256, 0, stream>>>(x, n1g, n1b, h16);

    // q = h @ Wq^T ; kv = h @ Wkv^T   (store f16; B tile via TDM->LDS)
    gemm_f16_kernel<0, true><<<dim3(ROWS / 128, DIM / 64), 256, 0, stream>>>(
        h16, wq16, nullptr, nullptr, q16, nullptr, ROWS, DIM, DIM);
    gemm_f16_kernel<0, true><<<dim3(ROWS / 128, DIM / 64), 256, 0, stream>>>(
        h16, wkv16, nullptr, nullptr, kv16, nullptr, ROWS, DIM, DIM);

    // linear attention (heads 1..7)
    lin_kstats<<<(BATCH * GHEADS * HD) / 8, 256, 0, stream>>>(kv16, kmax, ksinv);
    lin_ctx<<<BATCH * GHEADS, 1024, 0, stream>>>(kv16, kmax, ksinv, ctx);
    lin_out<<<(BATCH * GHEADS * SEQ) / 8, 256, 0, stream>>>(q16, ctx, attn16);

    // local attention (head 0)
    local_attn_kernel<<<BATCH * (SEQ / WINDOW), 256, 0, stream>>>(q16, kv16, attn16);

    // x2 = x + attn @ Wproj^T + bproj   (store f32; TDM B tile)
    gemm_f16_kernel<1, true><<<dim3(ROWS / 128, DIM / 64), 256, 0, stream>>>(
        attn16, wpr16, bproj, x, nullptr, x2, ROWS, DIM, DIM);

    // LN2
    ln_kernel<<<ROWS / 8, 256, 0, stream>>>(x2, n2g, n2b, h2_16);

    // mid = gelu(h2 @ W1^T + b1)   (store f16; TDM B tile, K=256)
    gemm_f16_kernel<2, true><<<dim3(ROWS / 128, MLPH / 64), 256, 0, stream>>>(
        h2_16, w1_16, b1, nullptr, mid16, nullptr, ROWS, MLPH, DIM);

    // out = x2 + mid @ W2^T + b2   (store f32; K=1024 -> direct-global B)
    gemm_f16_kernel<1, false><<<dim3(ROWS / 128, DIM / 64), 256, 0, stream>>>(
        mid16, w2_16, b2, x2, nullptr, out, ROWS, DIM, MLPH);
}
